// GraphAttentionLayer_13211319403263
// MI455X (gfx1250) — matrix operand order repaired and checked by reading, not compiled, not run on hardware
//
#include <hip/hip_runtime.h>
#include <hip/hip_bf16.h>
#include <cstdint>

// Problem constants (match reference)
#define NN 4096
#define CC 256
#define HH 8
#define DHH 32
#define EE 131072
#define EDD 16
#define SP (NN + 4)   // padded LDS row stride (floats): rows land on distinct banks
#define TP (CC + 4)   // padded LN tile stride

typedef __attribute__((ext_vector_type(16))) __bf16 v16bf;
typedef __attribute__((ext_vector_type(8)))  float  v8f;

union FragAB { v16bf v; unsigned u[8]; };

__device__ __forceinline__ unsigned short f2bf(float f) {
    unsigned u = __builtin_bit_cast(unsigned, f);
    u += 0x7FFFu + ((u >> 16) & 1u);          // round-to-nearest-even
    return (unsigned short)(u >> 16);
}
__device__ __forceinline__ unsigned pk_bf16(float lo, float hi) {
    return (unsigned)f2bf(lo) | ((unsigned)f2bf(hi) << 16);
}
__device__ __forceinline__ v8f vzero() {
    v8f z;
#pragma unroll
    for (int i = 0; i < 8; ++i) z[i] = 0.0f;
    return z;
}
// A-operand k order this code uses (contiguous k = 16h + s, matched by B): lanes 0-15 K=0..15, lanes 16-31 K=16..31, dword j = K{2j,2j+1}
__device__ __forceinline__ int a_k0(int j, int half) {
    return 2 * j + half * 16;
}
// B-operand k order this code uses (contiguous k = 16h + s, matched by A): lanes 0-15 K=0..15, lanes 16-31 K=16..31, dword j = K{2j,2j+1}
__device__ __forceinline__ int b_k0(int j, int half) {
    return 2 * j + half * 16;
}

// ---------------------------------------------------------------- conversions
__global__ void k_cvt_x(const float* __restrict__ x, unsigned short* __restrict__ xb) {
    int i = blockIdx.x * 256 + threadIdx.x;
    if (i < NN * CC) xb[i] = f2bf(x[i]);
}
// W stored (in,out) row-major; write transposed bf16 (out,in)
__global__ void k_cvt_wT(const float* __restrict__ w, unsigned short* __restrict__ wt) {
    int i = blockIdx.x * 256 + threadIdx.x;
    if (i < CC * CC) {
        int o = i >> 8, in = i & 255;
        wt[i] = f2bf(w[in * CC + o]);
    }
}
__global__ void k_zero(int* __restrict__ cnt, int* __restrict__ fill) {
    int i = blockIdx.x * 256 + threadIdx.x;
    if (i < NN) { cnt[i] = 0; fill[i] = 0; }
}

// ---------------------------------------------------------------- QKV GEMM
__global__ __launch_bounds__(128) void k_qkv(
    const unsigned short* __restrict__ xb,
    const unsigned short* __restrict__ wtq,
    const unsigned short* __restrict__ wtk,
    const unsigned short* __restrict__ wtv,
    const float* __restrict__ bq, const float* __restrict__ bk, const float* __restrict__ bv,
    unsigned short* __restrict__ qb,   // [H][N][DH]
    unsigned short* __restrict__ kb,   // [H][N][DH]
    unsigned short* __restrict__ vt)   // [H][DH][N]
{
    const int r0   = blockIdx.x * 16;
    const int lane = threadIdx.x & 31;
    const int wave = threadIdx.x >> 5;
    const int half = lane >> 4;
    const int n16  = lane & 15;

    v8f accq[4], acck[4], accv[4];
#pragma unroll
    for (int t = 0; t < 4; ++t) { accq[t] = vzero(); acck[t] = vzero(); accv[t] = vzero(); }

    for (int kc = 0; kc < CC; kc += 32) {
        FragAB a;
#pragma unroll
        for (int j = 0; j < 8; ++j)
            a.u[j] = *(const unsigned*)(xb + (size_t)(r0 + n16) * CC + kc + a_k0(j, half));
#pragma unroll
        for (int ct = 0; ct < 4; ++ct) {
            const int col = (wave * 4 + ct) * 16 + n16;
            FragAB b0, b1, b2;
#pragma unroll
            for (int j = 0; j < 8; ++j) {
                const int k0 = b_k0(j, half);
                b0.u[j] = *(const unsigned*)(wtq + (size_t)col * CC + kc + k0);
                b1.u[j] = *(const unsigned*)(wtk + (size_t)col * CC + kc + k0);
                b2.u[j] = *(const unsigned*)(wtv + (size_t)col * CC + kc + k0);
            }
            accq[ct] = __builtin_amdgcn_wmma_f32_16x16x32_bf16(false, a.v, false, b0.v, (short)0, accq[ct], false, false);
            acck[ct] = __builtin_amdgcn_wmma_f32_16x16x32_bf16(false, a.v, false, b1.v, (short)0, acck[ct], false, false);
            accv[ct] = __builtin_amdgcn_wmma_f32_16x16x32_bf16(false, a.v, false, b2.v, (short)0, accv[ct], false, false);
        }
    }
#pragma unroll
    for (int ct = 0; ct < 4; ++ct) {
        const int col = (wave * 4 + ct) * 16 + n16;
        const int h = col >> 5, dh = col & 31;
#pragma unroll
        for (int r = 0; r < 8; ++r) {
            const int row = r0 + r + half * 8;
            qb[((size_t)h * NN + row) * DHH + dh] = f2bf(accq[ct][r] + bq[col]);
            kb[((size_t)h * NN + row) * DHH + dh] = f2bf(acck[ct][r] + bk[col]);
            vt[((size_t)h * DHH + dh) * NN + row] = f2bf(accv[ct][r] + bv[col]);
        }
    }
}

// ---------------------------------------------------------------- edge pipeline
__global__ __launch_bounds__(256) void k_edge_vals(
    const long long* __restrict__ ei, const float* __restrict__ ea,
    const float* __restrict__ We, const float* __restrict__ be,
    float* __restrict__ evals, int* __restrict__ cnt)
{
    __shared__ float sw[EDD * HH];
    __shared__ float sb[HH];
    if (threadIdx.x < EDD * HH) sw[threadIdx.x] = We[threadIdx.x];
    if (threadIdx.x < HH) sb[threadIdx.x] = be[threadIdx.x];
    __syncthreads();
    const int e = blockIdx.x * 256 + threadIdx.x;
    if (e < EE) {
        float v[HH];
#pragma unroll
        for (int h = 0; h < HH; ++h) v[h] = sb[h];
#pragma unroll
        for (int d = 0; d < EDD; ++d) {
            const float a = ea[(size_t)e * EDD + d];
#pragma unroll
            for (int h = 0; h < HH; ++h) v[h] += a * sw[d * HH + h];
        }
#pragma unroll
        for (int h = 0; h < HH; ++h) evals[(size_t)e * HH + h] = v[h];
        atomicAdd(&cnt[(int)ei[e]], 1);
    }
}

__global__ __launch_bounds__(1024) void k_scan(const int* __restrict__ cnt, int* __restrict__ rowptr) {
    __shared__ int sc[1024];
    const int i = threadIdx.x;
    const int c0 = cnt[4 * i], c1 = cnt[4 * i + 1], c2 = cnt[4 * i + 2], c3 = cnt[4 * i + 3];
    const int s = c0 + c1 + c2 + c3;
    sc[i] = s;
    __syncthreads();
    for (int off = 1; off < 1024; off <<= 1) {
        const int v = (i >= off) ? sc[i - off] : 0;
        __syncthreads();
        sc[i] += v;
        __syncthreads();
    }
    const int excl = sc[i] - s;
    rowptr[4 * i]     = excl;
    rowptr[4 * i + 1] = excl + c0;
    rowptr[4 * i + 2] = excl + c0 + c1;
    rowptr[4 * i + 3] = excl + c0 + c1 + c2;
    if (i == 1023) rowptr[NN] = sc[1023];
}

__global__ __launch_bounds__(256) void k_scatter(
    const long long* __restrict__ ei, const float* __restrict__ evals,
    const int* __restrict__ rowptr, int* __restrict__ fill,
    int* __restrict__ ecol, float* __restrict__ esort)
{
    const int e = blockIdx.x * 256 + threadIdx.x;
    if (e < EE) {
        const int src = (int)ei[e];
        const int dst = (int)ei[EE + e];
        const int p = rowptr[src] + atomicAdd(&fill[src], 1);
        ecol[p] = dst;
#pragma unroll
        for (int h = 0; h < HH; ++h) esort[(size_t)p * HH + h] = evals[(size_t)e * HH + h];
    }
}

// ---------------------------------------------------------------- attention (score slab in LDS)
// 256 threads = 8 waves (2 waves/SIMD for latency hiding); LDS ~279 KB of the 320 KB/WGP.
__global__ __launch_bounds__(256) void k_attn(
    const unsigned short* __restrict__ qb,
    const unsigned short* __restrict__ kb,
    const unsigned short* __restrict__ vt,
    const int* __restrict__ rowptr, const int* __restrict__ ecol,
    const float* __restrict__ esort,
    unsigned short* __restrict__ ao)   // [N][C] bf16
{
    __shared__ float S[16 * SP];            // padded score/prob slab (bank-conflict-free rows)
    __shared__ float opart[8 * 16 * DHH];   // per-wave partial outputs
    __shared__ int   rp[17];

    const int h  = blockIdx.x >> 8;
    const int rb = blockIdx.x & 255;
    const int r0 = rb * 16;
    const int lane = threadIdx.x & 31;
    const int wave = threadIdx.x >> 5;
    const int half = lane >> 4;
    const int n16  = lane & 15;

    // ---- phase 1: S = (Q Kᵀ) / sqrt(DH)
    FragAB aq;
#pragma unroll
    for (int j = 0; j < 8; ++j)
        aq.u[j] = *(const unsigned*)(qb + ((size_t)h * NN + r0 + n16) * DHH + a_k0(j, half));
    const float scale = 0.17677669529663687f; // 1/sqrt(32)
    for (int ct = wave; ct < 256; ct += 8) {
        FragAB b;
#pragma unroll
        for (int j = 0; j < 8; ++j)
            b.u[j] = *(const unsigned*)(kb + ((size_t)h * NN + ct * 16 + n16) * DHH + b_k0(j, half));
        v8f d = vzero();
        d = __builtin_amdgcn_wmma_f32_16x16x32_bf16(false, aq.v, false, b.v, (short)0, d, false, false);
#pragma unroll
        for (int r = 0; r < 8; ++r)
            S[(r + half * 8) * SP + ct * 16 + n16] = d[r] * scale;
    }
    if (threadIdx.x < 17) rp[threadIdx.x] = rowptr[r0 + threadIdx.x];
    __syncthreads();

    // ---- phase 2: sparse edge bias scatter (CSR rows r0..r0+15 are contiguous)
    const int pbeg = rp[0], pend = rp[16];
    for (int p = pbeg + (int)threadIdx.x; p < pend; p += 256) {
        int m = 0;
        while (m < 15 && p >= rp[m + 1]) ++m;
        atomicAdd(&S[m * SP + ecol[p]], esort[(size_t)p * HH + h]);
    }
    __syncthreads();

    // ---- phase 3: softmax per row (16 lanes per row, shfl reductions intra-wave32)
    {
        const int m = threadIdx.x >> 4, sub = threadIdx.x & 15;
        float mx = -1e30f;
        for (int c = sub; c < NN; c += 16) mx = fmaxf(mx, S[m * SP + c]);
#pragma unroll
        for (int msk = 8; msk; msk >>= 1) mx = fmaxf(mx, __shfl_xor(mx, msk, 32));
        float sum = 0.0f;
        for (int c = sub; c < NN; c += 16) {
            const float e = __expf(S[m * SP + c] - mx);
            S[m * SP + c] = e;
            sum += e;
        }
#pragma unroll
        for (int msk = 8; msk; msk >>= 1) sum += __shfl_xor(sum, msk, 32);
        const float inv = 1.0f / sum;
        for (int c = sub; c < NN; c += 16) S[m * SP + c] *= inv;
    }
    __syncthreads();

    // ---- phase 4: O = P V, keys split across 8 waves
    v8f o0 = vzero(), o1 = vzero();
    const int kbeg = wave * 512;
    for (int kc = kbeg; kc < kbeg + 512; kc += 32) {
        FragAB a;
#pragma unroll
        for (int j = 0; j < 8; ++j) {
            const int k0 = a_k0(j, half);
            a.u[j] = pk_bf16(S[n16 * SP + kc + k0], S[n16 * SP + kc + k0 + 1]);
        }
        FragAB b0, b1;
#pragma unroll
        for (int j = 0; j < 8; ++j) {
            const int kk = kc + b_k0(j, half);
            b0.u[j] = *(const unsigned*)(vt + ((size_t)h * DHH + n16) * NN + kk);
            b1.u[j] = *(const unsigned*)(vt + ((size_t)h * DHH + 16 + n16) * NN + kk);
        }
        o0 = __builtin_amdgcn_wmma_f32_16x16x32_bf16(false, a.v, false, b0.v, (short)0, o0, false, false);
        o1 = __builtin_amdgcn_wmma_f32_16x16x32_bf16(false, a.v, false, b1.v, (short)0, o1, false, false);
    }
#pragma unroll
    for (int r = 0; r < 8; ++r) {
        opart[(wave * 16 + r + half * 8) * DHH + n16]      = o0[r];
        opart[(wave * 16 + r + half * 8) * DHH + 16 + n16] = o1[r];
    }
    __syncthreads();
    for (int i = threadIdx.x; i < 16 * DHH; i += 256) {
        const int m = i >> 5, dh = i & 31;
        float v = 0.0f;
#pragma unroll
        for (int w = 0; w < 8; ++w) v += opart[(w * 16 + m) * DHH + dh];
        ao[(size_t)(r0 + m) * CC + h * DHH + dh] = f2bf(v);
    }
}

// ---------------------------------------------------------------- output proj + residual + LN
__global__ __launch_bounds__(128) void k_out(
    const unsigned short* __restrict__ ao,
    const unsigned short* __restrict__ wto,
    const float* __restrict__ bo, const float* __restrict__ x,
    const float* __restrict__ gamma, const float* __restrict__ beta,
    float* __restrict__ out)
{
    __shared__ float T[16 * TP];
    const int r0   = blockIdx.x * 16;
    const int lane = threadIdx.x & 31;
    const int wave = threadIdx.x >> 5;
    const int half = lane >> 4;
    const int n16  = lane & 15;

    v8f acc[4];
#pragma unroll
    for (int t = 0; t < 4; ++t) acc[t] = vzero();

    for (int kc = 0; kc < CC; kc += 32) {
        FragAB a;
#pragma unroll
        for (int j = 0; j < 8; ++j)
            a.u[j] = *(const unsigned*)(ao + (size_t)(r0 + n16) * CC + kc + a_k0(j, half));
#pragma unroll
        for (int ct = 0; ct < 4; ++ct) {
            const int col = (wave * 4 + ct) * 16 + n16;
            FragAB b;
#pragma unroll
            for (int j = 0; j < 8; ++j)
                b.u[j] = *(const unsigned*)(wto + (size_t)col * CC + kc + b_k0(j, half));
            acc[ct] = __builtin_amdgcn_wmma_f32_16x16x32_bf16(false, a.v, false, b.v, (short)0, acc[ct], false, false);
        }
    }
#pragma unroll
    for (int ct = 0; ct < 4; ++ct) {
        const int col = (wave * 4 + ct) * 16 + n16;
#pragma unroll
        for (int r = 0; r < 8; ++r) {
            const int m = r + half * 8;
            T[m * TP + col] = acc[ct][r] + bo[col] + x[(size_t)(r0 + m) * CC + col];
        }
    }
    __syncthreads();
    const int m = threadIdx.x >> 3, sub = threadIdx.x & 7;
    float sum = 0.0f, sq = 0.0f;
    for (int c = sub; c < CC; c += 8) {
        const float v = T[m * TP + c];
        sum += v; sq += v * v;
    }
#pragma unroll
    for (int msk = 4; msk; msk >>= 1) {
        sum += __shfl_xor(sum, msk, 32);
        sq  += __shfl_xor(sq,  msk, 32);
    }
    const float mu  = sum * (1.0f / CC);
    const float var = sq * (1.0f / CC) - mu * mu;
    const float rin = rsqrtf(var + 1e-5f);
    for (int c = sub; c < CC; c += 8) {
        const float v = T[m * TP + c];
        out[(size_t)(r0 + m) * CC + c] = (v - mu) * rin * gamma[c] + beta[c];
    }
}

// ---------------------------------------------------------------- launch
static inline size_t alignup(size_t v) { return (v + 255) & ~(size_t)255; }

extern "C" void kernel_launch(void* const* d_in, const int* in_sizes, int n_in,
                              void* d_out, int out_size, void* d_ws, size_t ws_size,
                              hipStream_t stream) {
    const float*     x   = (const float*)d_in[0];
    const long long* ei  = (const long long*)d_in[1];
    const float*     ea  = (const float*)d_in[2];
    const float*     Wq  = (const float*)d_in[3];
    const float*     bq  = (const float*)d_in[4];
    const float*     Wk  = (const float*)d_in[5];
    const float*     bk  = (const float*)d_in[6];
    const float*     Wv  = (const float*)d_in[7];
    const float*     bv  = (const float*)d_in[8];
    const float*     We  = (const float*)d_in[9];
    const float*     be  = (const float*)d_in[10];
    const float*     Wo  = (const float*)d_in[11];
    const float*     bo  = (const float*)d_in[12];
    const float*     gm  = (const float*)d_in[13];
    const float*     bt  = (const float*)d_in[14];
    float* out = (float*)d_out;

    char* base = (char*)d_ws;
    size_t off = 0;
    unsigned short* xb  = (unsigned short*)(base + off); off += alignup((size_t)NN * CC * 2);
    unsigned short* wtq = (unsigned short*)(base + off); off += alignup((size_t)CC * CC * 2);
    unsigned short* wtk = (unsigned short*)(base + off); off += alignup((size_t)CC * CC * 2);
    unsigned short* wtv = (unsigned short*)(base + off); off += alignup((size_t)CC * CC * 2);
    unsigned short* wto = (unsigned short*)(base + off); off += alignup((size_t)CC * CC * 2);
    unsigned short* qb  = (unsigned short*)(base + off); off += alignup((size_t)HH * NN * DHH * 2);
    unsigned short* kb  = (unsigned short*)(base + off); off += alignup((size_t)HH * NN * DHH * 2);
    unsigned short* vt  = (unsigned short*)(base + off); off += alignup((size_t)HH * DHH * NN * 2);
    unsigned short* ao  = (unsigned short*)(base + off); off += alignup((size_t)NN * CC * 2);
    int*   cnt    = (int*)(base + off);   off += alignup((size_t)NN * 4);
    int*   fill   = (int*)(base + off);   off += alignup((size_t)NN * 4);
    int*   rowptr = (int*)(base + off);   off += alignup((size_t)(NN + 1) * 4);
    int*   ecolp  = (int*)(base + off);   off += alignup((size_t)EE * 4);
    float* evals  = (float*)(base + off); off += alignup((size_t)EE * HH * 4);
    float* esort  = (float*)(base + off); off += alignup((size_t)EE * HH * 4);

    k_cvt_x<<<(NN * CC + 255) / 256, 256, 0, stream>>>(x, xb);
    k_cvt_wT<<<(CC * CC + 255) / 256, 256, 0, stream>>>(Wq, wtq);
    k_cvt_wT<<<(CC * CC + 255) / 256, 256, 0, stream>>>(Wk, wtk);
    k_cvt_wT<<<(CC * CC + 255) / 256, 256, 0, stream>>>(Wv, wtv);
    k_cvt_wT<<<(CC * CC + 255) / 256, 256, 0, stream>>>(Wo, wto);
    k_zero<<<(NN + 255) / 256, 256, 0, stream>>>(cnt, fill);

    k_qkv<<<NN / 16, 128, 0, stream>>>(xb, wtq, wtk, wtv, bq, bk, bv, qb, kb, vt);

    k_edge_vals<<<EE / 256, 256, 0, stream>>>(ei, ea, We, be, evals, cnt);
    k_scan<<<1, 1024, 0, stream>>>(cnt, rowptr);
    k_scatter<<<EE / 256, 256, 0, stream>>>(ei, evals, rowptr, fill, ecolp, esort);

    k_attn<<<HH * (NN / 16), 256, 0, stream>>>(qb, kb, vt, rowptr, ecolp, esort, ao);

    k_out<<<NN / 16, 128, 0, stream>>>(ao, wto, bo, x, gm, bt, out);
}
